// MambaBlock_83743272338011
// MI455X (gfx1250) — compile-verified, matrix operand order checked by reading, not run on hardware
//
#include <hip/hip_runtime.h>
#include <hip/hip_bf16.h>

// ---------------------------------------------------------------------------
// Mamba block for MI455X (gfx1250), wave32 + WMMA bf16 path.
//   B=4, L=4096, D_MODEL=1024, D_INNER=2048, D_STATE=16, DT_RANK=64, D_CONV=4
//
// GEMM operands are pre-converted to bf16 (A row-major, B transposed to N x K)
// so every WMMA fragment is two contiguous global_load_b128 per lane
// (ISA 10.9: row-major 16-bit memory order == WMMA register layout).
// All GEMM control flow is compile-time (NTILES/FUSE/WRITE_BF templates) so
// accumulators stay pinned in place across the K-loop (no phi copies).
// ---------------------------------------------------------------------------

#define BSZ    4
#define LSEQ   4096
#define DMODEL 1024
#define DINNER 2048
#define NSTATE 16
#define DTRANK 64
#define XDBL_W (DTRANK + 2 * NSTATE)   // 96

typedef __attribute__((ext_vector_type(16))) __bf16 v16bf;
typedef __attribute__((ext_vector_type(8)))  __bf16 v8bf;
typedef __attribute__((ext_vector_type(8)))  float  v8f;

__device__ __forceinline__ float fast_sigmoid(float v) {
    return 1.0f / (1.0f + __expf(-v));
}
__device__ __forceinline__ float fast_softplus(float v) {
    return (v > 20.0f) ? v : __logf(1.0f + __expf(v));
}

// Load a 16-element bf16 fragment for one lane: two contiguous 16B loads at
// p (K=+0..7 relative) and p+16 (K=+16..23 relative).  Matches ISA 7.12.2
// 16-bit A/B layout when p = base + row*ld + k0 + 8*laneHalf.
__device__ __forceinline__ v16bf load_frag16(const __bf16* __restrict__ p) {
    v8bf lo = *reinterpret_cast<const v8bf*>(p);
    v8bf hi = *reinterpret_cast<const v8bf*>(p + 16);
    return __builtin_shufflevector(lo, hi, 0, 1, 2, 3, 4, 5, 6, 7,
                                           8, 9, 10, 11, 12, 13, 14, 15);
}

// ---------------------------------------------------------------------------
// WMMA GEMM:  C[M,N] = A[M,K] * Bt[N,K]^T  (+bias) (+softplus) (+bf16 copy)
//   A  : bf16 row-major, leading dim lda
//   Bt : bf16, B transposed (N x K), leading dim ldb
// Block = 4 waves; each wave computes a 32 x (16*NTILES) strip:
// 2 A-frags x NTILES B-frags = 2*NTILES v_wmma per 32-wide K-step.
// FUSE: 0 none, 1 +bias, 2 +bias,softplus.  No runtime branches anywhere.
// ---------------------------------------------------------------------------
template <int NTILES, int FUSE, bool WRITE_BF>
__global__ void __launch_bounds__(128)
gemm_wmma_bf16(const __bf16* __restrict__ A, int lda,
               const __bf16* __restrict__ Bt, int ldb,
               const float* __restrict__ bias,
               float* __restrict__ C, int ldc,
               __bf16* __restrict__ Cbf,
               int K)
{
    const int lane = threadIdx.x;            // 0..31
    const int wave = threadIdx.y;            // 0..3
    const int nn   = lane & 15;              // row/col index within 16-tile
    const int hi   = lane >> 4;              // lane half -> K split
    const int mt   = (blockIdx.y * 4 + wave) * 32;     // 32 rows per wave
    const int nt   = blockIdx.x * (16 * NTILES);       // col-strip base

    const size_t arow0 = (size_t)(mt + nn) * lda;
    const size_t arow1 = (size_t)(mt + 16 + nn) * lda;

    const v8f zero = {0.f, 0.f, 0.f, 0.f, 0.f, 0.f, 0.f, 0.f};
    v8f acc0[NTILES], acc1[NTILES];
#pragma unroll
    for (int t = 0; t < NTILES; ++t) { acc0[t] = zero; acc1[t] = zero; }

    for (int k0 = 0; k0 < K; k0 += 32) {
        const int ko = k0 + (hi << 3);
        const v16bf a0 = load_frag16(A + arow0 + ko);
        const v16bf a1 = load_frag16(A + arow1 + ko);
#pragma unroll
        for (int t = 0; t < NTILES; ++t) {
            const v16bf b =
                load_frag16(Bt + (size_t)(nt + 16 * t + nn) * ldb + ko);
            acc0[t] = __builtin_amdgcn_wmma_f32_16x16x32_bf16(
                false, a0, false, b, (short)0, acc0[t], false, false);
            acc1[t] = __builtin_amdgcn_wmma_f32_16x16x32_bf16(
                false, a1, false, b, (short)0, acc1[t], false, false);
        }
    }

    // Epilogue.  C/D layout: VGPR j -> row (j + 8*hi), col = lane&15.
#pragma unroll
    for (int t = 0; t < NTILES; ++t) {
        const int col = nt + 16 * t + nn;
        const float bv = (FUSE >= 1) ? bias[col] : 0.0f;
#pragma unroll
        for (int j = 0; j < 8; ++j) {
            {
                const int row = mt + j + 8 * hi;
                float v = acc0[t][j] + bv;
                if (FUSE == 2) v = fast_softplus(v);
                C[(size_t)row * ldc + col] = v;
                if (WRITE_BF) Cbf[(size_t)row * ldc + col] = (__bf16)v;
            }
            {
                const int row = mt + 16 + j + 8 * hi;
                float v = acc1[t][j] + bv;
                if (FUSE == 2) v = fast_softplus(v);
                C[(size_t)row * ldc + col] = v;
                if (WRITE_BF) Cbf[(size_t)row * ldc + col] = (__bf16)v;
            }
        }
    }
}

// ---------------------------------------------------------------------------
// fp32 -> bf16 elementwise convert (for the input activations x).
// ---------------------------------------------------------------------------
__global__ void __launch_bounds__(256)
cvt_bf16_kernel(const float* __restrict__ in, __bf16* __restrict__ out, size_t n)
{
    const size_t idx = (size_t)blockIdx.x * blockDim.x + threadIdx.x;
    if (idx < n) out[idx] = (__bf16)in[idx];
}

// ---------------------------------------------------------------------------
// Transpose + convert:  W[K,N] fp32  ->  Wt[N,K] bf16 (coalesced writes).
// ---------------------------------------------------------------------------
__global__ void __launch_bounds__(256)
transpose_cvt_kernel(const float* __restrict__ W, __bf16* __restrict__ Wt,
                     int K, int N)
{
    const size_t idx = (size_t)blockIdx.x * blockDim.x + threadIdx.x;
    if (idx >= (size_t)N * K) return;
    const int n = (int)(idx / K);
    const int k = (int)(idx % K);
    Wt[idx] = (__bf16)W[(size_t)k * N + n];
}

// ---------------------------------------------------------------------------
// Causal depthwise conv (K=4, left pad 3) + bias + SiLU.
// Writes xc in fp32 (scan input) and bf16 (GEMM-3 A operand).
// ---------------------------------------------------------------------------
__global__ void __launch_bounds__(256)
conv_silu_kernel(const float* __restrict__ xar,
                 const float* __restrict__ conv_w,
                 const float* __restrict__ conv_b,
                 float* __restrict__ xc,
                 __bf16* __restrict__ xc_bf)
{
    const size_t idx = (size_t)blockIdx.x * blockDim.x + threadIdx.x;
    const int d = (int)(idx % DINNER);
    const size_t tok = idx / DINNER;           // b*L + l
    const int l = (int)(tok % LSEQ);

    float acc = conv_b[d];
#pragma unroll
    for (int k = 0; k < 4; ++k) {
        const int ll = l - 3 + k;
        if (ll >= 0) {
            acc += conv_w[d * 4 + k] *
                   xar[(tok + (size_t)(ll - l)) * (2 * DINNER) + d];
        }
    }
    const float s = acc * fast_sigmoid(acc);
    xc[idx] = s;
    xc_bf[idx] = (__bf16)s;
}

// ---------------------------------------------------------------------------
// Selective scan.  One lane per (b, d) channel: 16-state recurrence in
// registers, sequential over L.  B_ssm/C_ssm broadcast from x_dbl rows
// (6.3 MB -> resident in the 192 MB L2).  Fused +D*x and *silu(res);
// output written directly as bf16 (GEMM-6 A operand).
// ---------------------------------------------------------------------------
__global__ void __launch_bounds__(256)
scan_kernel(const float* __restrict__ delta,   // [B,L,DINNER] post-softplus
            const float* __restrict__ xc,      // [B,L,DINNER]
            const float* __restrict__ xdbl,    // [B,L,96]
            const float* __restrict__ xar,     // [B,L,2*DINNER] (res half)
            const float* __restrict__ A_log,   // [DINNER,16]
            const float* __restrict__ D_par,   // [DINNER]
            __bf16* __restrict__ yz_bf)        // [B,L,DINNER]
{
    const int idx = blockIdx.x * blockDim.x + threadIdx.x;   // 0..B*DINNER-1
    const int d = idx % DINNER;
    const int b = idx / DINNER;

    float Aval[NSTATE];
#pragma unroll
    for (int n = 0; n < NSTATE; ++n)
        Aval[n] = -__expf(A_log[d * NSTATE + n]);
    const float Dp = D_par[d];

    float h[NSTATE];
#pragma unroll
    for (int n = 0; n < NSTATE; ++n) h[n] = 0.0f;

    for (int l = 0; l < LSEQ; ++l) {
        const size_t tok = (size_t)b * LSEQ + l;
        const float dt = delta[tok * DINNER + d];
        const float xt = xc[tok * DINNER + d];

        float4 Bq[4], Cq[4];
        const float4* bp =
            reinterpret_cast<const float4*>(xdbl + tok * XDBL_W + DTRANK);
        const float4* cp =
            reinterpret_cast<const float4*>(xdbl + tok * XDBL_W + DTRANK + NSTATE);
#pragma unroll
        for (int i = 0; i < 4; ++i) { Bq[i] = bp[i]; Cq[i] = cp[i]; }
        const float* Bv = reinterpret_cast<const float*>(Bq);
        const float* Cv = reinterpret_cast<const float*>(Cq);

        float y = 0.0f;
#pragma unroll
        for (int n = 0; n < NSTATE; ++n) {
            const float dA = __expf(dt * Aval[n]);
            h[n] = fmaf(h[n], dA, (dt * Bv[n]) * xt);
            y = fmaf(h[n], Cv[n], y);
        }
        y = fmaf(xt, Dp, y);

        const float r = xar[tok * (2 * DINNER) + DINNER + d];
        yz_bf[tok * DINNER + d] = (__bf16)(y * (r * fast_sigmoid(r)));
    }
}

// ---------------------------------------------------------------------------
extern "C" void kernel_launch(void* const* d_in, const int* in_sizes, int n_in,
                              void* d_out, int out_size, void* d_ws, size_t ws_size,
                              hipStream_t stream)
{
    (void)in_sizes; (void)n_in; (void)out_size; (void)ws_size;

    const float* x      = (const float*)d_in[0];   // [B,L,DMODEL]
    const float* W_in   = (const float*)d_in[1];   // [DMODEL, 2*DINNER]
    const float* b_in   = (const float*)d_in[2];
    const float* conv_w = (const float*)d_in[3];   // [DINNER,1,4]
    const float* conv_b = (const float*)d_in[4];
    const float* W_x    = (const float*)d_in[5];   // [DINNER, 96]
    const float* W_dt   = (const float*)d_in[6];   // [DTRANK, DINNER]
    const float* b_dt   = (const float*)d_in[7];
    const float* A_log  = (const float*)d_in[8];   // [DINNER,16]
    const float* D_par  = (const float*)d_in[9];
    const float* W_out  = (const float*)d_in[10];  // [DINNER, DMODEL]
    const float* b_out  = (const float*)d_in[11];
    float* out = (float*)d_out;                    // [B,L,DMODEL]

    const int M = BSZ * LSEQ;                      // 16384 tokens

    // ---- workspace carve (256B aligned slabs) ----
    char* cur = (char*)d_ws;
    auto carve = [&](size_t bytes) {
        void* p = cur;
        cur += (bytes + 255) & ~(size_t)255;
        return p;
    };
    float*  xar     = (float*) carve((size_t)M * 2 * DINNER * 4); // x_and_res
    float*  xc      = (float*) carve((size_t)M * DINNER * 4);
    float*  xdbl    = (float*) carve((size_t)M * XDBL_W * 4);
    float*  delta   = (float*) carve((size_t)M * DINNER * 4);
    __bf16* x_bf    = (__bf16*)carve((size_t)M * DMODEL * 2);
    __bf16* xc_bf   = (__bf16*)carve((size_t)M * DINNER * 2);
    __bf16* xdbl_bf = (__bf16*)carve((size_t)M * XDBL_W * 2);
    __bf16* yz_bf   = (__bf16*)carve((size_t)M * DINNER * 2);
    __bf16* Wt_in   = (__bf16*)carve((size_t)DMODEL * 2 * DINNER * 2);
    __bf16* Wt_x    = (__bf16*)carve((size_t)DINNER * XDBL_W * 2);
    __bf16* Wt_dt   = (__bf16*)carve((size_t)DTRANK * DINNER * 2);
    __bf16* Wt_out  = (__bf16*)carve((size_t)DINNER * DMODEL * 2);

    const dim3 blk(32, 4);   // 4 waves (wave32)

    // ---- operand preparation (bf16 / transposed-bf16) ----
    {
        const size_t n = (size_t)M * DMODEL;
        cvt_bf16_kernel<<<(unsigned)((n + 255) / 256), 256, 0, stream>>>(x, x_bf, n);
    }
    {
        const size_t n = (size_t)(2 * DINNER) * DMODEL;        // Wt_in [4096,1024]
        transpose_cvt_kernel<<<(unsigned)((n + 255) / 256), 256, 0, stream>>>(
            W_in, Wt_in, DMODEL, 2 * DINNER);
    }
    {
        const size_t n = (size_t)XDBL_W * DINNER;              // Wt_x [96,2048]
        transpose_cvt_kernel<<<(unsigned)((n + 255) / 256), 256, 0, stream>>>(
            W_x, Wt_x, DINNER, XDBL_W);
    }
    {
        const size_t n = (size_t)DINNER * DTRANK;              // Wt_dt [2048,64]
        transpose_cvt_kernel<<<(unsigned)((n + 255) / 256), 256, 0, stream>>>(
            W_dt, Wt_dt, DTRANK, DINNER);
    }
    {
        const size_t n = (size_t)DMODEL * DINNER;              // Wt_out [1024,2048]
        transpose_cvt_kernel<<<(unsigned)((n + 255) / 256), 256, 0, stream>>>(
            W_out, Wt_out, DINNER, DMODEL);
    }

    // 1) x_and_res = x @ W_in + b_in        [16384,1024] x [1024,4096]
    {
        dim3 grid((2 * DINNER) / 64, M / 128);    // 64-wide strips
        gemm_wmma_bf16<4, 1, false><<<grid, blk, 0, stream>>>(
            x_bf, DMODEL, Wt_in, DMODEL, b_in, xar, 2 * DINNER, nullptr, DMODEL);
    }

    // 2) xc = silu(depthwise_conv(x_in) + conv_b)   (fp32 + bf16)
    {
        const int total = M * DINNER;              // 33.5M
        conv_silu_kernel<<<total / 256, 256, 0, stream>>>(
            xar, conv_w, conv_b, xc, xc_bf);
    }

    // 3) x_dbl = xc @ W_x                   [16384,2048] x [2048,96]
    {
        dim3 grid(XDBL_W / 48, M / 128);          // 48-wide strips, no tail
        gemm_wmma_bf16<3, 0, true><<<grid, blk, 0, stream>>>(
            xc_bf, DINNER, Wt_x, DINNER, nullptr, xdbl, XDBL_W, xdbl_bf, DINNER);
    }

    // 4) delta = softplus(d_r @ W_dt + b_dt)  [16384,64] x [64,2048]
    {
        dim3 grid(DINNER / 64, M / 128);
        gemm_wmma_bf16<4, 2, false><<<grid, blk, 0, stream>>>(
            xdbl_bf, XDBL_W, Wt_dt, DTRANK, b_dt, delta, DINNER, nullptr, DTRANK);
    }

    // 5) selective scan -> yz (bf16), fused +D*x and *silu(res)
    {
        scan_kernel<<<(BSZ * DINNER) / 256, 256, 0, stream>>>(
            delta, xc, xdbl, xar, A_log, D_par, yz_bf);
    }

    // 6) out = yz @ W_out + b_out           [16384,2048] x [2048,1024]
    {
        dim3 grid(DMODEL / 64, M / 128);
        gemm_wmma_bf16<4, 1, false><<<grid, blk, 0, stream>>>(
            yz_bf, DINNER, Wt_out, DINNER, b_out, out, DMODEL, nullptr, DINNER);
    }
}
